// VIDEOSWIN_15281493639780
// MI455X (gfx1250) — compile-verified, matrix operand order checked by reading, not run on hardware
//
#include <hip/hip_runtime.h>
#include <hip/hip_bf16.h>

// Problem constants (Video Swin window attention)
#define TOK   98          // tokens per window
#define TOKP  112         // padded tokens = 7 * 16
#define CD    128         // channels
#define HD    32          // head dim
#define NH_   4           // heads
#define NWIN  256         // windows per batch image (mask count)
#define ROWS  200704      // B_ * N = 2048 * 98
#define MT    (ROWS / 16) // 12544 M-tiles
#define SSTR  116         // f32 score row stride (pad vs 112 for banking)

typedef __attribute__((ext_vector_type(16))) _Float16 v16h;
typedef __attribute__((ext_vector_type(8)))  float    v8f;

union HFrag { v16h v; _Float16 e[16]; };
union FAcc  { v8f  v; float    e[8];  };

// ---- CDNA5 async global->LDS copy (16 B per lane, ASYNCcnt-tracked) ----
// GVS addressing: mem_addr = SADDR(u64) + VADDR(u32); LDS dest addr in VDST VGPR.
__device__ inline void async_ld_b128(void* ldsp, unsigned long long gbase, unsigned voffB) {
  unsigned lds = (unsigned)(unsigned long long)ldsp; // low 32 bits of LDS aperture addr
  asm volatile("global_load_async_to_lds_b128 %0, %1, %2"
               :: "v"(lds), "v"(voffB), "s"(gbase)
               : "memory");
}
__device__ inline void wait_async0() {
  asm volatile("s_wait_asynccnt 0" ::: "memory");
}

// ---- WMMA fragment loaders (wave32 layouts per CDNA5 ISA 7.12.2) ----
// A (16x32 f16): lane&15 = row; lanes<16 hold K {0..7,16..23}, lanes>=16 hold {8..15,24..31}
// B (32x16 f16): symmetric with "row" -> column of B (i.e. N-major source).
__device__ inline v16h frag_ld_f16(const _Float16* p, int stride, int r, int k0, int kb) {
  HFrag f;
  const _Float16* q = p + (size_t)r * stride + k0 + kb;
#pragma unroll
  for (int i = 0; i < 8; ++i) { f.e[i] = q[i]; f.e[8 + i] = q[16 + i]; }
  return f.v;
}

__device__ inline v16h frag_ld_f32(const float* p, int stride, int r, int k0, int kb) {
  HFrag f;
  const float* q = p + (size_t)r * stride + k0 + kb;
#pragma unroll
  for (int i = 0; i < 8; ++i) { f.e[i] = (_Float16)q[i]; f.e[8 + i] = (_Float16)q[16 + i]; }
  return f.v;
}

// B fragment from a row-major K x N source (element [k][n] at p[k*stride + n])
__device__ inline v16h frag_ld_f16_rm(const _Float16* p, int stride, int col, int k0, int kb) {
  HFrag f;
  const _Float16* q = p + (size_t)(k0 + kb) * stride + col;
#pragma unroll
  for (int i = 0; i < 8; ++i) { f.e[i] = q[i * stride]; f.e[8 + i] = q[(16 + i) * stride]; }
  return f.v;
}

// ---- Kernel A: qkv = x @ qkv_w^T + qkv_b, stored f16 (ROWS x 384) ----
__global__ __launch_bounds__(256) void vswin_qkv_gemm(
    const float* __restrict__ x, const float* __restrict__ w,
    const float* __restrict__ bias, _Float16* __restrict__ qkv)
{
  const int lane  = threadIdx.x & 31;
  const int l16   = lane & 15;
  const int kb    = (lane & 16) ? 8 : 0;
  const int mtile = blockIdx.x * 8 + (threadIdx.x >> 5);
  if (mtile >= MT) return;
  const int arow = mtile * 16 + l16;

  v16h a[4];
#pragma unroll
  for (int kc = 0; kc < 4; ++kc) a[kc] = frag_ld_f32(x, CD, arow, kc * 32, kb);

  for (int nt = 0; nt < 24; ++nt) {
    const int ncol = nt * 16 + l16;
    if (nt + 1 < 24) __builtin_prefetch(w + (size_t)(ncol + 16) * CD, 0, 3);
    FAcc acc{};
#pragma unroll
    for (int kc = 0; kc < 4; ++kc) {
      v16h b = frag_ld_f32(w, CD, ncol, kc * 32, kb);   // B[k,n] = W[n,k]
      acc.v = __builtin_amdgcn_wmma_f32_16x16x32_f16(false, a[kc], false, b,
                                                     (short)0, acc.v, false, false);
    }
    const float bv = bias[ncol];
#pragma unroll
    for (int i = 0; i < 8; ++i)
      qkv[(size_t)(mtile * 16 + kb + i) * 384 + ncol] = (_Float16)(acc.e[i] + bv);
  }
}

// ---- Kernel B: fused window attention per (window b, head h) ----
__global__ __launch_bounds__(256) void vswin_attn(
    const _Float16* __restrict__ qkv, const float* __restrict__ rpb,
    const int* __restrict__ relidx, const float* __restrict__ mask,
    _Float16* __restrict__ ao)
{
  __shared__ __align__(16) _Float16 qs[TOKP * HD];   // 112 x 32
  __shared__ __align__(16) _Float16 ks[TOKP * HD];   // 112 x 32
  __shared__ __align__(16) _Float16 vs[CD * HD];     // 128 x 32 (K-padded for PV gemm)
  __shared__ float    sS[TOKP * SSTR];               // f32 scores
  __shared__ __align__(16) _Float16 sP[TOKP * CD];   // 112 x 128 probabilities (K-padded)

  const int b    = blockIdx.x >> 2;
  const int h    = blockIdx.x & 3;
  const int tid  = threadIdx.x;
  const int lane = tid & 31;
  const int wv   = tid >> 5;
  const int l16  = lane & 15;
  const int kb   = (lane & 16) ? 8 : 0;
  const float scale = 0.17677669529663687f; // 32^-0.5 (applied to scores)

  const unsigned long long base =
      (unsigned long long)(qkv + (size_t)b * TOK * 384 + h * HD);
  const uint4 zz = {0u, 0u, 0u, 0u};

  // Stage q,k (112x32) and v (128x32) into LDS via async b128 copies;
  // zero-fill the padded rows with vector LDS stores.
  for (int i = tid; i < TOKP * 4; i += 256) {      // 4 x 16B chunks per 32-half row
    const int r = i >> 2, c = (i & 3) * 8;
    if (r < TOK) {
      const unsigned voff = (unsigned)(r * 384 + c) * 2u;
      async_ld_b128(&qs[r * HD + c], base, voff);
      async_ld_b128(&ks[r * HD + c], base, voff + (unsigned)(CD * 2));
    } else {
      *reinterpret_cast<uint4*>(&qs[r * HD + c]) = zz;
      *reinterpret_cast<uint4*>(&ks[r * HD + c]) = zz;
    }
  }
  for (int i = tid; i < CD * 4; i += 256) {
    const int r = i >> 2, c = (i & 3) * 8;
    if (r < TOK)
      async_ld_b128(&vs[r * HD + c], base, (unsigned)(r * 384 + 2 * CD + c) * 2u);
    else
      *reinterpret_cast<uint4*>(&vs[r * HD + c]) = zz;
  }
  wait_async0();
  __syncthreads();

  // S = scale * (q @ k^T) : 7x7 = 49 tiles, K = hd = 32 -> one WMMA each
  for (int t = wv; t < 49; t += 8) {
    const int mt = t / 7, nt = t % 7;
    v16h aq = frag_ld_f16(qs, HD, mt * 16 + l16, 0, kb);
    v16h bk = frag_ld_f16(ks, HD, nt * 16 + l16, 0, kb); // B[k,n] = K[n,k]
    FAcc acc{};
    acc.v = __builtin_amdgcn_wmma_f32_16x16x32_f16(false, aq, false, bk,
                                                   (short)0, acc.v, false, false);
#pragma unroll
    for (int i = 0; i < 8; ++i)
      sS[(mt * 16 + kb + i) * SSTR + nt * 16 + l16] = acc.e[i] * scale;
  }
  __syncthreads();

  // Bias gather + shift mask + softmax, one row per thread
  const float* maskw = mask + (size_t)(b & (NWIN - 1)) * TOK * TOK;
  for (int r = tid; r < TOKP; r += 256) {
    _Float16* prow = &sP[r * CD];
    if (r < TOK) {
      const int*   ri = relidx + r * TOK;
      const float* mk = maskw + r * TOK;
      float* srow = &sS[r * SSTR];
      float m = -1e30f;
      for (int c = 0; c < TOK; ++c) {
        float v = srow[c] + rpb[ri[c] * NH_ + h] + mk[c];
        srow[c] = v;
        m = fmaxf(m, v);
      }
      float sum = 0.f;
      for (int c = 0; c < TOK; ++c) {
        float e = __expf(srow[c] - m);
        srow[c] = e;
        sum += e;
      }
      const float inv = 1.f / sum;
      for (int c = 0; c < TOK; ++c) prow[c] = (_Float16)(srow[c] * inv);
      for (int c = TOK; c < CD; ++c) prow[c] = (_Float16)0.f; // padded K cols
    } else {
      for (int c = 0; c < CD; ++c) prow[c] = (_Float16)0.f;   // padded rows
    }
  }
  __syncthreads();

  // O = P @ V : 7 m-tiles x 2 n-tiles, K = 128 -> 4 WMMA each
  for (int t = wv; t < 14; t += 8) {
    const int mt = t >> 1, nt = (t & 1) * 16;
    FAcc acc{};
#pragma unroll
    for (int kc = 0; kc < 4; ++kc) {
      v16h ap = frag_ld_f16(sP, CD, mt * 16 + l16, kc * 32, kb);
      v16h bv = frag_ld_f16_rm(vs, HD, nt + l16, kc * 32, kb); // V row-major KxN
      acc.v = __builtin_amdgcn_wmma_f32_16x16x32_f16(false, ap, false, bv,
                                                     (short)0, acc.v, false, false);
    }
#pragma unroll
    for (int i = 0; i < 8; ++i) {
      const int r = mt * 16 + kb + i;
      if (r < TOK)
        ao[(size_t)(b * TOK + r) * CD + h * HD + nt + l16] = (_Float16)acc.e[i];
    }
  }
}

// ---- Kernel C: out = ao @ proj_w^T + proj_b (f32 output) ----
__global__ __launch_bounds__(256) void vswin_proj_gemm(
    const _Float16* __restrict__ ao, const float* __restrict__ w,
    const float* __restrict__ bias, float* __restrict__ out)
{
  const int lane  = threadIdx.x & 31;
  const int l16   = lane & 15;
  const int kb    = (lane & 16) ? 8 : 0;
  const int mtile = blockIdx.x * 8 + (threadIdx.x >> 5);
  if (mtile >= MT) return;
  const int arow = mtile * 16 + l16;

  v16h a[4];
#pragma unroll
  for (int kc = 0; kc < 4; ++kc) a[kc] = frag_ld_f16(ao, CD, arow, kc * 32, kb);

  for (int nt = 0; nt < 8; ++nt) {
    const int ncol = nt * 16 + l16;
    if (nt + 1 < 8) __builtin_prefetch(w + (size_t)(ncol + 16) * CD, 0, 3);
    FAcc acc{};
#pragma unroll
    for (int kc = 0; kc < 4; ++kc) {
      v16h bw = frag_ld_f32(w, CD, ncol, kc * 32, kb);   // B[k,n] = W[n,k]
      acc.v = __builtin_amdgcn_wmma_f32_16x16x32_f16(false, a[kc], false, bw,
                                                     (short)0, acc.v, false, false);
    }
    const float bv = bias[ncol];
#pragma unroll
    for (int i = 0; i < 8; ++i)
      out[(size_t)(mtile * 16 + kb + i) * CD + ncol] = acc.e[i] + bv;
  }
}

extern "C" void kernel_launch(void* const* d_in, const int* in_sizes, int n_in,
                              void* d_out, int out_size, void* d_ws, size_t ws_size,
                              hipStream_t stream) {
  const float* x        = (const float*)d_in[0];
  const float* qkv_w    = (const float*)d_in[1];
  const float* qkv_b    = (const float*)d_in[2];
  const float* rpb      = (const float*)d_in[3];
  const float* proj_w   = (const float*)d_in[4];
  const float* proj_b   = (const float*)d_in[5];
  const int*   relidx   = (const int*)d_in[6];
  const float* mask     = (const float*)d_in[7];
  float*       out      = (float*)d_out;

  _Float16* qkvbuf = (_Float16*)d_ws;                       // ROWS x 384 f16
  _Float16* aobuf  = qkvbuf + (size_t)ROWS * 384;           // ROWS x 128 f16

  vswin_qkv_gemm<<<MT / 8, 256, 0, stream>>>(x, qkv_w, qkv_b, qkvbuf);
  vswin_attn<<<2048 * NH_, 256, 0, stream>>>(qkvbuf, rpb, relidx, mask, aobuf);
  vswin_proj_gemm<<<MT / 8, 256, 0, stream>>>(aobuf, proj_w, proj_b, out);
}
